// IterBlock_2track_no_perceiver_188978561517
// MI455X (gfx1250) — matrix operand order repaired and checked by reading, not compiled, not run on hardware
//
#include <hip/hip_runtime.h>

// ---------------------------------------------------------------------------
// CDNA5 (gfx1250, wave32) Evoformer iter-block.
// All contractions go through v_wmma_f32_16x16x32_bf16 via a generic
// strided/batched GEMM kernel (bf16 LDS staging, f32 accumulate).
// Block tile: 32(M) x 64(N), K-step 32; 4 waves, each wave = 16x32 C
// (one A fragment reused across two WMMAs).
// ---------------------------------------------------------------------------

typedef __attribute__((ext_vector_type(16))) __bf16        v16bf;
typedef __attribute__((ext_vector_type(8)))  float         v8f;
typedef __attribute__((ext_vector_type(8)))  unsigned int  v8u;

// ------------------------------- GEMM --------------------------------------

struct GemmP {
  const float* A; const float* B; const float* bias; const float* emul; const float* resid;
  float* C;
  int M, N, K, K2, B2;
  long sAb1, sAb2, sAm, sAk1, sAk2;
  long sBb1, sBb2, sBn, sBk1, sBk2;
  long sCb1, sCb2, sCm, sCn;
  float scale; int act;
};

__device__ __forceinline__ unsigned short f2bf(float f) {
  unsigned x = __float_as_uint(f);
  return (unsigned short)((x + 0x7FFFu + ((x >> 16) & 1u)) >> 16);   // RNE
}

__device__ __forceinline__ float actf(float v, int act) {
  if (act == 1) return v > 0.f ? v : 0.f;                 // relu
  if (act == 2) return 1.f / (1.f + __expf(-v));          // sigmoid
  if (act == 3) return v > 0.f ? v : (__expf(v) - 1.f);   // elu
  return v;
}

__device__ __forceinline__ void store_c(const GemmP& p, long offC, int m, int n,
                                        float v) {
  if (p.bias) v += p.bias[n];
  v *= p.scale;
  v = actf(v, p.act);
  const long ci = offC + (long)m * p.sCm + (long)n * p.sCn;
  if (p.emul)  v *= p.emul[ci];
  if (p.resid) v += p.resid[ci];
  p.C[ci] = v;
}

// Generic batched GEMM:
//   bz -> (b1,b2) = (bz/B2, bz%B2)
//   kappa -> (k1,k2) = (kappa/K2, kappa%K2)   [hoisted: once per K-step]
//   C[...] = resid + emul * act( (sum_k A*B + bias[n]) * scale )
__launch_bounds__(128)
__global__ void k_gemm(GemmP p) {
  __shared__ unsigned short Ash[32 * 32];   // [m][k]
  __shared__ unsigned short Bsh[64 * 32];   // [n][k]  (transposed stage)
  const int bz = blockIdx.z;
  const int b1 = bz / p.B2, b2 = bz % p.B2;
  const long offA = (long)b1 * p.sAb1 + (long)b2 * p.sAb2;
  const long offB = (long)b1 * p.sBb1 + (long)b2 * p.sBb2;
  const long offC = (long)b1 * p.sCb1 + (long)b2 * p.sCb2;
  const int tileM = blockIdx.y * 32;
  const int tileN = blockIdx.x * 64;
  const int tid = threadIdx.x;
  const int lane = tid & 31, wv = tid >> 5;
  const int wm = (wv >> 1) * 16, wn = (wv & 1) * 32;
  const int kk = tid & 31;        // constant K-column per thread for staging
  const int rbase = tid >> 5;     // 0..3

  v8f acc0 = {}, acc1 = {};
  for (unsigned kt = 0; kt < (unsigned)p.K; kt += 32) {
    // ---- per-thread K offset, computed once per K-step (hoisted div/mod) --
    const unsigned kap = kt + (unsigned)kk;
    const bool kok = kap < (unsigned)p.K;
    long aKoff = 0, bKoff = 0;
    if (kok) {
      const unsigned k1 = kap / (unsigned)p.K2, k2 = kap % (unsigned)p.K2;
      aKoff = (long)k1 * p.sAk1 + (long)k2 * p.sAk2;
      bKoff = (long)k1 * p.sBk1 + (long)k2 * p.sBk2;
    }
    const float* Abase = p.A + offA + aKoff;
    const float* Bbase = p.B + offB + bKoff;

    // speculative prefetch of next K-step (exact when K is contiguous)
    if (kt + 32 < (unsigned)p.K) {
      __builtin_prefetch(Abase + 32 * p.sAk2 + (long)(tileM + rbase) * p.sAm, 0, 1);
      __builtin_prefetch(Bbase + 32 * p.sBk2 + (long)(tileN + rbase) * p.sBn, 0, 1);
    }

#pragma unroll
    for (int r = 0; r < 8; ++r) {           // A tile: 32 rows
      const int mr = rbase + 4 * r;
      const int m = tileM + mr;
      const float v = (kok && m < p.M) ? Abase[(long)m * p.sAm] : 0.f;
      Ash[mr * 32 + kk] = f2bf(v);
    }
#pragma unroll
    for (int r = 0; r < 16; ++r) {          // B tile: 64 n-rows
      const int nr = rbase + 4 * r;
      const int n = tileN + nr;
      const float v = (kok && n < p.N) ? Bbase[(long)n * p.sBn] : 0.f;
      Bsh[nr * 32 + kk] = f2bf(v);
    }
    __syncthreads();

    // ---- assemble WMMA fragments from LDS (ISA 16-bit layouts) -----------
    const int g = lane >> 4, rr = lane & 15;
    const unsigned* A32 = (const unsigned*)Ash;
    const unsigned* B32 = (const unsigned*)Bsh;
    v8u au, bu0, bu1;
    const int arow = wm + rr;
#pragma unroll
    for (int vv = 0; vv < 8; ++vv) {
      const int kb = (vv < 4) ? (2 * vv + 8 * g) : (2 * (vv - 4) + 16 + 8 * g);
      au[vv] = A32[(arow * 32 + kb) >> 1];
    }
    const int brow0 = wn + rr, brow1 = wn + 16 + rr;
#pragma unroll
    for (int t = 0; t < 8; ++t) {
      const int kb = (t < 4) ? (2 * t + 8 * g) : (2 * (t - 4) + 16 + 8 * g);
      bu0[t] = B32[(brow0 * 32 + kb) >> 1];
      bu1[t] = B32[(brow1 * 32 + kb) >> 1];
    }
    v16bf av  = __builtin_bit_cast(v16bf, au);
    v16bf bv0 = __builtin_bit_cast(v16bf, bu0);
    v16bf bv1 = __builtin_bit_cast(v16bf, bu1);
    acc0 = __builtin_amdgcn_wmma_f32_16x16x32_bf16(false, av, false, bv0,
                                                   (short)0, acc0, false, false);
    acc1 = __builtin_amdgcn_wmma_f32_16x16x32_bf16(false, av, false, bv1,
                                                   (short)0, acc1, false, false);
    __syncthreads();
  }

  // ---- epilogue: C layout M = r + 8*(lane>>4), N = lane&15 ----------------
  const int g = lane >> 4;
  const int n0 = tileN + wn + (lane & 15);
  const int n1 = n0 + 16;
#pragma unroll
  for (int r = 0; r < 8; ++r) {
    const int m = tileM + wm + r + 8 * g;
    if (m < p.M) {
      if (n0 < p.N) store_c(p, offC, m, n0, acc0[r]);
      if (n1 < p.N) store_c(p, offC, m, n1, acc1[r]);
    }
  }
}

// ----------------------------- LayerNorm -----------------------------------

__launch_bounds__(128)
__global__ void k_layernorm(const float* x, const float* gg, const float* bb,
                            float* y, int d, float eps) {
  const long row = blockIdx.x;
  const float* xr = x + row * (long)d;
  float* yr = y + row * (long)d;
  __shared__ float red[128];
  float s = 0.f;
  for (int i = threadIdx.x; i < d; i += 128) s += xr[i];
  red[threadIdx.x] = s; __syncthreads();
  for (int o = 64; o > 0; o >>= 1) {
    if ((int)threadIdx.x < o) red[threadIdx.x] += red[threadIdx.x + o];
    __syncthreads();
  }
  const float mu = red[0] / d; __syncthreads();
  float s2 = 0.f;
  for (int i = threadIdx.x; i < d; i += 128) { float t = xr[i] - mu; s2 += t * t; }
  red[threadIdx.x] = s2; __syncthreads();
  for (int o = 64; o > 0; o >>= 1) {
    if ((int)threadIdx.x < o) red[threadIdx.x] += red[threadIdx.x + o];
    __syncthreads();
  }
  const float rstd = rsqrtf(red[0] / d + eps);
  for (int i = threadIdx.x; i < d; i += 128)
    yr[i] = (xr[i] - mu) * rstd * gg[i] + bb[i];
}

// ------------------------------ Softmax ------------------------------------
// row base = (r/rdiv)*rs1 + (r%rdiv)*rs2 ; element stride `stride`
// bias base = ((r/bd1)%bm1)*bs1 + ((r/bd2)%bm2)*bs2 ; element stride bstride

struct SoftP {
  float* x; const float* bias;
  int nrows, len;
  long stride, rdiv, rs1, rs2;
  long bd1, bm1, bs1, bd2, bm2, bs2, bstride;
};

__launch_bounds__(128)
__global__ void k_softmax(SoftP p) {
  const int r = blockIdx.x * 4 + (threadIdx.x >> 5);
  const int lane = threadIdx.x & 31;
  if (r >= p.nrows) return;
  const long base = ((long)r / p.rdiv) * p.rs1 + ((long)r % p.rdiv) * p.rs2;
  long bbase = 0;
  if (p.bias)
    bbase = (((long)r / p.bd1) % p.bm1) * p.bs1 + (((long)r / p.bd2) % p.bm2) * p.bs2;
  float mx = -3.4e38f;
  for (int i = lane; i < p.len; i += 32) {
    float v = p.x[base + (long)i * p.stride];
    if (p.bias) v += p.bias[bbase + (long)i * p.bstride];
    mx = fmaxf(mx, v);
  }
  for (int o = 16; o > 0; o >>= 1) mx = fmaxf(mx, __shfl_xor(mx, o, 32));
  float sum = 0.f;
  for (int i = lane; i < p.len; i += 32) {
    float v = p.x[base + (long)i * p.stride];
    if (p.bias) v += p.bias[bbase + (long)i * p.bstride];
    sum += __expf(v - mx);
  }
  for (int o = 16; o > 0; o >>= 1) sum += __shfl_xor(sum, o, 32);
  const float inv = 1.f / sum;
  for (int i = lane; i < p.len; i += 32) {
    float v = p.x[base + (long)i * p.stride];
    if (p.bias) v += p.bias[bbase + (long)i * p.bstride];
    p.x[base + (long)i * p.stride] = __expf(v - mx) * inv;
  }
}

// --------------------------- small kernels ---------------------------------

__global__ void k_ew_mul(const float* a, const float* b, float* c, long n) {
  long i = (long)blockIdx.x * 256 + threadIdx.x;
  if (i < n) c[i] = a[i] * b[i];
}
__global__ void k_mul_bcast8(float* q, const float* sw, long n) {
  long i = (long)blockIdx.x * 256 + threadIdx.x;
  if (i < n) q[i] *= sw[i >> 3];
}
__global__ void k_colgate(const float* g, const float* co, float* o, long n) {
  long i = (long)blockIdx.x * 256 + threadIdx.x;
  if (i < n) o[i] = g[i] * co[i % 12288];   // 12288 = L*hm
}
__global__ void k_swdot(const float* q0, const float* ksw, float* sw,
                        float scale, long n) {
  long i = (long)blockIdx.x * 256 + threadIdx.x;
  if (i >= n) return;
  const int h = (int)(i & 7);
  const long sl = i >> 3;
  const int l = (int)(sl % 192);
  const float* qp = q0 + (long)l * 64 + h * 8;
  const float* kp = ksw + i * 8;
  float s = 0.f;
#pragma unroll
  for (int d = 0; d < 8; ++d) s += qp[d] * kp[d];
  sw[i] = s * scale;
}
__global__ void k_mean_s(const float* q, float* ql, float scale) {
  long i = (long)blockIdx.x * 256 + threadIdx.x;
  if (i >= 12288) return;
  float s = 0.f;
  for (int ss = 0; ss < 128; ++ss) s += q[(long)ss * 12288 + i];
  ql[i] = s * scale;
}
__global__ void k_add_elu(const float* a, const float* b, float* c, long n) {
  long i = (long)blockIdx.x * 256 + threadIdx.x;
  if (i < n) { float v = a[i] + b[i]; c[i] = v > 0.f ? v : (__expf(v) - 1.f); }
}
__global__ void k_transpose_pair(const float* in, float* out, long n) {
  long i = (long)blockIdx.x * 256 + threadIdx.x;
  if (i >= n) return;
  const int c = (int)(i % 128);
  const long ij = i / 128;
  const int j = (int)(ij % 192);
  const int ii = (int)(ij / 192);
  out[((long)j * 192 + ii) * 128 + c] = in[i];
}
__global__ void k_pad(const float* in, float* out, long n) {  // 194x194x128
  long i = (long)blockIdx.x * 256 + threadIdx.x;
  if (i >= n) return;
  const int c = (int)(i % 128);
  const long r = i / 128;
  const int x = (int)(r % 194);
  const int y = (int)(r / 194);
  float v = 0.f;
  if (x >= 1 && x <= 192 && y >= 1 && y <= 192)
    v = in[((long)(y - 1) * 192 + (x - 1)) * 128 + c];
  out[i] = v;
}
__global__ void k_instat(const float* x, float* stat, float eps) {
  const int c = blockIdx.x;  // 128 channels
  __shared__ float r1[256], r2[256];
  float s = 0.f, s2 = 0.f;
  for (long p = threadIdx.x; p < 36864; p += 256) {
    const float v = x[p * 128 + c]; s += v; s2 += v * v;
  }
  r1[threadIdx.x] = s; r2[threadIdx.x] = s2; __syncthreads();
  for (int o = 128; o > 0; o >>= 1) {
    if ((int)threadIdx.x < o) { r1[threadIdx.x] += r1[threadIdx.x + o]; r2[threadIdx.x] += r2[threadIdx.x + o]; }
    __syncthreads();
  }
  if (threadIdx.x == 0) {
    const float mu = r1[0] / 36864.f;
    const float var = r2[0] / 36864.f - mu * mu;
    stat[c] = mu; stat[128 + c] = rsqrtf(var + eps);
  }
}
__global__ void k_inapply(float* x, const float* stat, const float* g,
                          const float* b, long n, int act) {
  long i = (long)blockIdx.x * 256 + threadIdx.x;
  if (i >= n) return;
  const int c = (int)(i & 127);
  float v = (x[i] - stat[c]) * stat[128 + c] * g[c] + b[c];
  if (act == 3) v = v > 0.f ? v : (__expf(v) - 1.f);
  x[i] = v;
}

// --------------------------- host launchers --------------------------------

static void launch_gemm(hipStream_t st, const float* A, const float* B,
                        const float* bias, const float* emul, const float* resid,
                        float* C, int M, int N, int K, int K2, int nbatch, int B2,
                        long sAb1, long sAb2, long sAm, long sAk1, long sAk2,
                        long sBb1, long sBb2, long sBn, long sBk1, long sBk2,
                        long sCb1, long sCb2, long sCm, long sCn,
                        float scale, int act) {
  GemmP p;
  p.A = A; p.B = B; p.bias = bias; p.emul = emul; p.resid = resid; p.C = C;
  p.M = M; p.N = N; p.K = K; p.K2 = K2; p.B2 = B2;
  p.sAb1 = sAb1; p.sAb2 = sAb2; p.sAm = sAm; p.sAk1 = sAk1; p.sAk2 = sAk2;
  p.sBb1 = sBb1; p.sBb2 = sBb2; p.sBn = sBn; p.sBk1 = sBk1; p.sBk2 = sBk2;
  p.sCb1 = sCb1; p.sCb2 = sCb2; p.sCm = sCm; p.sCn = sCn;
  p.scale = scale; p.act = act;
  dim3 grid((N + 63) / 64, (M + 31) / 32, nbatch);
  k_gemm<<<grid, dim3(128, 1, 1), 0, st>>>(p);
}

static void lin(hipStream_t st, const float* A, const float* W, const float* bias,
                float* C, int M, int N, int K, float scale = 1.f, int act = 0,
                const float* emul = nullptr, const float* resid = nullptr) {
  launch_gemm(st, A, W, bias, emul, resid, C, M, N, K, K, 1, 1,
              0, 0, (long)K, 0, 1,
              0, 0, 1, 0, (long)N,
              0, 0, (long)N, 1, scale, act);
}

static void softmaxrows(hipStream_t st, float* x, const float* bias, int nrows,
                        int len, long stride, long rdiv, long rs1, long rs2,
                        long bd1, long bm1, long bs1, long bd2, long bm2,
                        long bs2, long bstride) {
  SoftP p;
  p.x = x; p.bias = bias; p.nrows = nrows; p.len = len;
  p.stride = stride; p.rdiv = rdiv; p.rs1 = rs1; p.rs2 = rs2;
  p.bd1 = bd1; p.bm1 = bm1; p.bs1 = bs1; p.bd2 = bd2; p.bm2 = bm2; p.bs2 = bs2;
  p.bstride = bstride;
  k_softmax<<<(nrows + 3) / 4, 128, 0, st>>>(p);
}

static inline int nb(long n) { return (int)((n + 255) / 256); }

// ------------------------------ pipeline -----------------------------------

extern "C" void kernel_launch(void* const* d_in, const int* in_sizes, int n_in,
                              void* d_out, int out_size, void* d_ws, size_t ws_size,
                              hipStream_t stream) {
  auto PF = [&](int i) { return (const float*)d_in[i]; };
  // d_in flattened pytree (alphabetical): 0 msa, 1 pair, then params:
  // col:2-13  ff_msa:14-19  ff_pair:20-25  m2p:26-41  row:42-61
  // ta_e:62-75  ta_s:76-89  tm_in:90-105  tm_out:106-121
  const float* msa_in  = PF(0);
  const float* pair_in = PF(1);
  float* out_msa  = (float*)d_out;              // 128*192*256
  float* out_pair = out_msa + 6291456L;         // 192*192*128

  float* w = (float*)d_ws;
  long off = 0;
  auto alloc = [&](long n) { float* p = w + off; off += n; return p; };

  const long MSA_E = 6291456L, PAIR_E = 4718592L;
  float* mbuf   = alloc(MSA_E);
  float* zbuf   = alloc(PAIR_E);
  float* msaA   = alloc(MSA_E);
  float* msaB   = alloc(MSA_E);
  float* pairA  = alloc(PAIR_E);
  float* pairB  = alloc(PAIR_E);
  float* big    = alloc(37748736L);             // op / tri-attn logits / ffn h1
  float* qbuf   = alloc(1572864L);
  float* kbuf   = alloc(1572864L);
  float* vbuf   = alloc(1572864L);
  float* gatebuf= alloc(1572864L);
  float* outbuf = alloc(1572864L);
  float* kswbuf = alloc(1572864L);
  float* swbuf  = alloc(196608L);
  float* logitsbuf = alloc(294912L);
  float* biasbuf   = alloc(294912L);
  float* q0buf  = alloc(12288L);
  float* qlbuf  = alloc(12288L);
  float* cl_logits = alloc(196608L);
  float* coloutbuf = alloc(12288L);
  float* leftbuf  = alloc(1179648L);
  float* rightbuf = alloc(1179648L);
  float* tmbuf    = alloc(1179648L);
  float* tmln     = alloc(1179648L);
  float* oplbuf   = alloc(PAIR_E);
  float* prbuf    = alloc(PAIR_E);
  float* padbuf   = alloc(4818432L);            // 194*194*128
  float* conv1out = alloc(PAIR_E);
  float* conv2out = alloc(PAIR_E);
  float* pairQ = alloc(PAIR_E);
  float* pairK = alloc(PAIR_E);
  float* pairV = alloc(PAIR_E);
  float* pairG = alloc(PAIR_E);
  float* pairO = alloc(PAIR_E);
  float* biaszbuf = alloc(147456L);
  float* instat   = alloc(256L);

  const float INV_SQRT8  = 0.35355339059327373f;
  const float INV_SQRT32 = 0.17677669529663687f;

  // ===================== row attention (soft-tied, pair bias) ==============
  k_layernorm<<<24576, 128, 0, stream>>>(msa_in, PF(49), PF(48), mbuf, 256, 1e-5f);
  k_layernorm<<<36864, 128, 0, stream>>>(pair_in, PF(51), PF(50), zbuf, 128, 1e-5f);
  lin(stream, mbuf, PF(59), PF(58), q0buf, 192, 64, 256);                 // sw_q on seq 0
  lin(stream, mbuf, PF(57), PF(56), kswbuf, 24576, 64, 256);              // sw_k
  k_swdot<<<nb(196608), 256, 0, stream>>>(q0buf, kswbuf, swbuf, INV_SQRT8, 196608);
  softmaxrows(stream, swbuf, nullptr, 1536, 128, 1536, 1536, 0, 1,
              1, 1, 0, 1, 1, 0, 0);                                       // softmax over s
  lin(stream, mbuf, PF(55), PF(54), qbuf, 24576, 64, 256);
  lin(stream, mbuf, PF(47), PF(46), kbuf, 24576, 64, 256, INV_SQRT8);
  lin(stream, mbuf, PF(61), PF(60), vbuf, 24576, 64, 256);
  k_mul_bcast8<<<nb(1572864), 256, 0, stream>>>(qbuf, swbuf, 1572864);
  lin(stream, zbuf, PF(43), PF(42), biasbuf, 36864, 8, 128);              // pair bias
  // attn[i,j,h] = sum_{s,d} q*k  (K=(s,d), batch=h)
  launch_gemm(stream, qbuf, kbuf, nullptr, nullptr, nullptr, logitsbuf,
              192, 192, 1024, 8, 8, 8,
              0, 8, 64, 12288, 1,
              0, 8, 64, 12288, 1,
              0, 1, 1536, 8, 1.f, 0);
  softmaxrows(stream, logitsbuf, biasbuf, 1536, 192, 8, 8, 1536, 1,
              8, 1 << 28, 1536, 1, 8, 1, 8);                              // softmax over j
  // out[s,i,h,d] = attn[i,:,h] x v[s,:,h,:]  (batch (s,h))
  launch_gemm(stream, logitsbuf, vbuf, nullptr, nullptr, nullptr, outbuf,
              192, 8, 192, 192, 1024, 8,
              0, 1, 1536, 0, 8,
              12288, 8, 1, 0, 64,
              12288, 8, 64, 1, 1.f, 0);
  lin(stream, mbuf, PF(45), PF(44), gatebuf, 24576, 64, 256, 1.f, 2);
  k_ew_mul<<<nb(1572864), 256, 0, stream>>>(outbuf, gatebuf, outbuf, 1572864);
  lin(stream, outbuf, PF(53), PF(52), msaA, 24576, 256, 64, 1.f, 0, nullptr, msa_in);

  // ===================== column global attention ===========================
  k_layernorm<<<24576, 128, 0, stream>>>(msaA, PF(7), PF(6), mbuf, 256, 1e-5f);
  lin(stream, mbuf, PF(11), PF(10), qbuf, 24576, 64, 256);
  k_mean_s<<<nb(12288), 256, 0, stream>>>(qbuf, qlbuf, 0.0027621358640644486f); // 1/(128*sqrt8)
  lin(stream, mbuf, PF(5), PF(4), kbuf, 24576, 8, 256);
  lin(stream, mbuf, PF(13), PF(12), vbuf, 24576, 8, 256);
  launch_gemm(stream, qlbuf, kbuf, nullptr, nullptr, nullptr, cl_logits,
              8, 128, 8, 8, 192, 1,
              64, 0, 8, 0, 1,
              8, 0, 1536, 0, 1,
              1024, 0, 128, 1, 1.f, 0);
  softmaxrows(stream, cl_logits, nullptr, 1536, 128, 1, 1, 128, 0,
              1, 1, 0, 1, 1, 0, 0);                                       // over s
  launch_gemm(stream, cl_logits, vbuf, nullptr, nullptr, nullptr, coloutbuf,
              8, 8, 128, 128, 192, 1,
              1024, 0, 128, 0, 1,
              8, 0, 1, 0, 1536,
              64, 0, 8, 1, 1.f, 0);
  lin(stream, mbuf, PF(3), PF(2), gatebuf, 24576, 64, 256, 1.f, 2);
  k_colgate<<<nb(1572864), 256, 0, stream>>>(gatebuf, coloutbuf, outbuf, 1572864);
  lin(stream, outbuf, PF(9), PF(8), msaB, 24576, 256, 64, 1.f, 0, nullptr, msaA);

  // ===================== MSA feed-forward ==================================
  k_layernorm<<<24576, 128, 0, stream>>>(msaB, PF(19), PF(18), mbuf, 256, 1e-5f);
  lin(stream, mbuf, PF(15), PF(14), big, 24576, 1024, 256, 1.f, 1);
  lin(stream, big, PF(17), PF(16), out_msa, 24576, 256, 1024, 1.f, 0, nullptr, msaB);

  // ===================== MSA -> pair =======================================
  k_layernorm<<<24576, 128, 0, stream>>>(out_msa, PF(37), PF(36), mbuf, 256, 1e-5f);
  lin(stream, mbuf, PF(35), PF(34), leftbuf, 24576, 32, 256);
  lin(stream, mbuf, PF(41), PF(40), rightbuf, 24576, 32, 256, 0.0078125f); // /N
  // op[l,i,(m,j)] = sum_s left[s,l,i] right[s,m,j]   (batch l)
  launch_gemm(stream, leftbuf, rightbuf, nullptr, nullptr, nullptr, big,
              32, 6144, 128, 128, 192, 1,
              32, 0, 1, 0, 6144,
              0, 0, 1, 0, 6144,
              196608, 0, 6144, 1, 1.f, 0);
  // op_lin[l,m,c] = sum_{i,j} op * W[(i,j),c]  (batch l)
  launch_gemm(stream, big, PF(39), PF(38), nullptr, nullptr, oplbuf,
              192, 128, 1024, 32, 192, 1,
              196608, 0, 32, 6144, 1,
              0, 0, 1, 4096, 128,
              24576, 0, 128, 1, 1.f, 0);
  // pr = concat[pair, op] @ down + b  (split into two accumulating GEMMs)
  launch_gemm(stream, pair_in, PF(29), nullptr, nullptr, nullptr, prbuf,
              36864, 128, 128, 128, 1, 1,
              0, 0, 128, 0, 1,
              0, 0, 1, 0, 128,
              0, 0, 128, 1, 1.f, 0);
  launch_gemm(stream, oplbuf, PF(29) + 128 * 128, PF(28), nullptr, prbuf, prbuf,
              36864, 128, 128, 128, 1, 1,
              0, 0, 128, 0, 1,
              0, 0, 1, 0, 128,
              0, 0, 128, 1, 1.f, 0);
  // residual conv block: conv1 -> IN -> ELU -> conv2 -> IN; elu(pr + h)
  auto conv9 = [&](const float* pad, const float* cw, float* dst) {
    for (int t = 0; t < 9; ++t) {
      const int dy = t / 3, dx = t % 3;
      launch_gemm(stream, pad + (long)(dy * 194 + dx) * 128, cw + (long)t * 16384,
                  nullptr, nullptr, (t == 0 ? nullptr : dst), dst,
                  192, 128, 128, 128, 192, 1,
                  24832, 0, 128, 0, 1,
                  0, 0, 1, 0, 128,
                  24576, 0, 128, 1, 1.f, 0);
    }
  };
  k_pad<<<nb(4818432), 256, 0, stream>>>(prbuf, padbuf, 4818432);
  conv9(padbuf, PF(26), conv1out);
  k_instat<<<128, 256, 0, stream>>>(conv1out, instat, 1e-6f);
  k_inapply<<<nb(PAIR_E), 256, 0, stream>>>(conv1out, instat, PF(31), PF(30), PAIR_E, 3);
  k_pad<<<nb(4818432), 256, 0, stream>>>(conv1out, padbuf, 4818432);
  conv9(padbuf, PF(27), conv2out);
  k_instat<<<128, 256, 0, stream>>>(conv2out, instat, 1e-6f);
  k_inapply<<<nb(PAIR_E), 256, 0, stream>>>(conv2out, instat, PF(33), PF(32), PAIR_E, 0);
  k_add_elu<<<nb(PAIR_E), 256, 0, stream>>>(prbuf, conv2out, pairA, PAIR_E);

  // ===================== triangle multiplication ===========================
  auto trimul = [&](const float* pin, float* pout, int bs, bool outgoing) {
    k_layernorm<<<36864, 128, 0, stream>>>(pin, PF(bs + 7), PF(bs + 6), zbuf, 128, 1e-5f);
    lin(stream, zbuf, PF(bs + 3), PF(bs + 2), tmln, 36864, 32, 128, 1.f, 2);        // sig(lg)
    lin(stream, zbuf, PF(bs + 5), PF(bs + 4), leftbuf, 36864, 32, 128, 1.f, 0, tmln, nullptr);
    lin(stream, zbuf, PF(bs + 13), PF(bs + 12), tmln, 36864, 32, 128, 1.f, 2);      // sig(rg)
    lin(stream, zbuf, PF(bs + 15), PF(bs + 14), rightbuf, 36864, 32, 128, 1.f, 0, tmln, nullptr);
    if (outgoing)  // out[i,j,d] = sum_k L[i,k,d] R[j,k,d]
      launch_gemm(stream, leftbuf, rightbuf, nullptr, nullptr, nullptr, tmbuf,
                  192, 192, 192, 192, 32, 1,
                  1, 0, 6144, 0, 32,
                  1, 0, 6144, 0, 32,
                  1, 0, 6144, 32, 1.f, 0);
    else           // out[i,j,d] = sum_k L[k,i,d] R[k,j,d]
      launch_gemm(stream, leftbuf, rightbuf, nullptr, nullptr, nullptr, tmbuf,
                  192, 192, 192, 192, 32, 1,
                  1, 0, 32, 0, 6144,
                  1, 0, 32, 0, 6144,
                  1, 0, 6144, 32, 1.f, 0);
    k_layernorm<<<36864, 128, 0, stream>>>(tmbuf, PF(bs + 9), PF(bs + 8), tmln, 32, 1e-5f);
    lin(stream, zbuf, PF(bs + 1), PF(bs + 0), pairG, 36864, 128, 128, 1.f, 2);      // gate
    lin(stream, tmln, PF(bs + 11), PF(bs + 10), pout, 36864, 128, 32, 1.f, 0, pairG, pin);
  };
  trimul(pairA, pairB, 106, true);   // tm_out
  trimul(pairB, pairA, 90, false);   // tm_in

  // ===================== triangle attention ================================
  auto triattn = [&](const float* pin, float* pout, int bs) {
    k_layernorm<<<36864, 128, 0, stream>>>(pin, PF(bs + 7), PF(bs + 6), zbuf, 128, 1e-5f);
    lin(stream, zbuf, PF(bs + 11), PF(bs + 10), pairQ, 36864, 128, 128, INV_SQRT32);
    lin(stream, zbuf, PF(bs + 5), PF(bs + 4), pairK, 36864, 128, 128);
    lin(stream, zbuf, PF(bs + 13), PF(bs + 12), pairV, 36864, 128, 128);
    lin(stream, zbuf, PF(bs + 3), PF(bs + 2), pairG, 36864, 128, 128, 1.f, 2);
    lin(stream, zbuf, PF(bs + 1), PF(bs + 0), biaszbuf, 36864, 4, 128);
    // logits[i,h,j,k] = sum_d q[i,j,h,d] k[i,k,h,d]
    launch_gemm(stream, pairQ, pairK, nullptr, nullptr, nullptr, big,
                192, 192, 32, 32, 768, 4,
                24576, 32, 128, 0, 1,
                24576, 32, 128, 0, 1,
                147456, 36864, 192, 1, 1.f, 0);
    softmaxrows(stream, big, biaszbuf, 147456, 192, 1, 1, 192, 0,
                1, 192, 768, 192, 4, 1, 4);                               // over k, +bias[j,k,h]
    // out[i,j,h,d] = sum_k attn[i,h,j,k] v[i,k,h,d]
    launch_gemm(stream, big, pairV, nullptr, nullptr, nullptr, pairO,
                192, 32, 192, 192, 768, 4,
                147456, 36864, 192, 0, 1,
                24576, 32, 1, 0, 128,
                24576, 32, 128, 1, 1.f, 0);
    k_ew_mul<<<nb(PAIR_E), 256, 0, stream>>>(pairO, pairG, pairO, PAIR_E);
    lin(stream, pairO, PF(bs + 9), PF(bs + 8), pout, 36864, 128, 128, 1.f, 0, nullptr, pin);
  };
  triattn(pairA, pairB, 76);                                               // ta_s
  k_transpose_pair<<<nb(PAIR_E), 256, 0, stream>>>(pairB, conv1out, PAIR_E);
  triattn(conv1out, conv2out, 62);                                         // ta_e (transposed)
  k_transpose_pair<<<nb(PAIR_E), 256, 0, stream>>>(conv2out, pairA, PAIR_E);

  // ===================== pair feed-forward =================================
  k_layernorm<<<36864, 128, 0, stream>>>(pairA, PF(25), PF(24), zbuf, 128, 1e-5f);
  lin(stream, zbuf, PF(21), PF(20), big, 36864, 256, 128, 1.f, 1);
  lin(stream, big, PF(23), PF(22), out_pair, 36864, 128, 256, 1.f, 0, nullptr, pairA);

  (void)in_sizes; (void)n_in; (void)out_size; (void)ws_size;
}
